// InterpretableMultiHeadAttention_35046933135750
// MI455X (gfx1250) — matrix operand order repaired and checked by reading, not compiled, not run on hardware
//
#include <hip/hip_runtime.h>

typedef __bf16 bf16_t;
typedef __attribute__((ext_vector_type(16))) __bf16 v16bf;
typedef __attribute__((ext_vector_type(8)))  __bf16 v8bf;
typedef __attribute__((ext_vector_type(4)))  __bf16 v4bf;
typedef __attribute__((ext_vector_type(8)))  float   v8f;

#define BATCH  4
#define SEQ    2048
#define DMODEL 1024

// ---------------------------------------------------------------------------
// Uniform bf16 WMMA GEMM:  C = scale * (A @ B^T) + bias
//   A: [M][K] bf16 row-major, B: [N][K] bf16 row-major
//   OUT_F32: write fp32, else bf16.  OUT_TR: write C transposed [N][M].
// Block: 128 threads = 4 waves (2x2), block tile 128x128, K-step 64.
// Each wave owns a 64x64 output tile: per 32-K sub-step it preloads
// 4 A-frags + 4 B-frags (16 ds_load_b128) then issues 16 WMMAs back-to-back.
// Staging: global->LDS via GLOBAL_LOAD_ASYNC_TO_LDS_B128 (ASYNCcnt),
// double-buffered so tile t+1 streams while tile t feeds the WMMAs.
// ---------------------------------------------------------------------------
template<bool OUT_F32, bool OUT_TR, bool HAS_BIAS>
__global__ __launch_bounds__(128)
void gemm_wmma_bf16(const bf16_t* __restrict__ A,
                    const bf16_t* __restrict__ Bm,
                    const float* __restrict__ bias,
                    void* __restrict__ Cptr,
                    int K, int lda, int ldb, int ldc,
                    long sA, long sB, long sC, float scale)
{
    constexpr int BM = 128, BN = 128, BK = 64;
    constexpr int LSTR = BK + 8;                 // padded row (144 B), 16B aligned
    __shared__ __align__(16) bf16_t As[2][BM][LSTR];
    __shared__ __align__(16) bf16_t Bs[2][BN][LSTR];

    const int m0   = blockIdx.y * BM;
    const int n0   = blockIdx.x * BN;
    const int tid  = threadIdx.x;
    const int wave = tid >> 5;                   // 0..3
    const int lane = tid & 31;
    const int wm   = (wave >> 1) * 64;           // 0,64
    const int wn   = (wave & 1) * 64;            // 0,64
    const int l15  = lane & 15;

    const bf16_t* Ab = A  + (size_t)blockIdx.z * sA;
    const bf16_t* Bb = Bm + (size_t)blockIdx.z * sB;

    const int srow = tid >> 3;                   // 0..15
    const int scol = (tid & 7) * 8;              // 0..56 (16-byte segments)

    // Issue 16 async global->LDS 16B copies (8 rows of A + 8 rows of B).
    auto stage = [&](int buf, int kt) {
        #pragma unroll
        for (int rr = srow; rr < BM; rr += 16) {
            const bf16_t* ga = Ab + (size_t)(m0 + rr) * lda + kt + scol;
            unsigned la = (unsigned)(size_t)&As[buf][rr][scol];
            asm volatile("global_load_async_to_lds_b128 %0, %1, off"
                         :: "v"(la), "v"((unsigned long long)(size_t)ga) : "memory");
            const bf16_t* gb = Bb + (size_t)(n0 + rr) * ldb + kt + scol;
            unsigned lb = (unsigned)(size_t)&Bs[buf][rr][scol];
            asm volatile("global_load_async_to_lds_b128 %0, %1, off"
                         :: "v"(lb), "v"((unsigned long long)(size_t)gb) : "memory");
        }
    };

    v8f acc[4][4] = {};
    const int akb = (lane < 16) ? 0 : 8;         // A frag K split {0..7,16..23}/{8..15,24..31}
    const int bkb = (lane < 16) ? 0 : 8;         // B frag K split {0..7,16..23}/{8..15,24..31}

    stage(0, 0);
    const int nk = K / BK;
    for (int t = 0; t < nk; ++t) {
        const int cur = t & 1;
        if (t + 1 < nk) {
            stage(cur ^ 1, (t + 1) * BK);
            asm volatile("s_wait_asynccnt 0x10" ::: "memory");  // only next tile in flight
        } else {
            asm volatile("s_wait_asynccnt 0x0" ::: "memory");
        }
        __syncthreads();

        #pragma unroll
        for (int ks = 0; ks < BK; ks += 32) {
            union Frag { v16bf v; v8bf h[2]; };
            Frag afr[4], bfr[4];
            #pragma unroll
            for (int mt = 0; mt < 4; ++mt) {
                const int row = wm + mt * 16 + l15;
                afr[mt].h[0] = *(const v8bf*)&As[cur][row][ks + akb];
                afr[mt].h[1] = *(const v8bf*)&As[cur][row][ks + akb + 16];
            }
            #pragma unroll
            for (int nt = 0; nt < 4; ++nt) {
                const int col = wn + nt * 16 + l15;
                bfr[nt].h[0] = *(const v8bf*)&Bs[cur][col][ks + bkb];
                bfr[nt].h[1] = *(const v8bf*)&Bs[cur][col][ks + bkb + 16];
            }
            #pragma unroll
            for (int mt = 0; mt < 4; ++mt)
                #pragma unroll
                for (int nt = 0; nt < 4; ++nt)
                    acc[mt][nt] = __builtin_amdgcn_wmma_f32_16x16x32_bf16(
                        false, afr[mt].v, false, bfr[nt].v, (short)0,
                        acc[mt][nt], false, false);
        }
        __syncthreads();   // all waves done reading `cur` before it is restaged
    }

    // Epilogue: VGPR r -> M = r (lanes<16) / r+8 (lanes>=16), N = lane%16
    float*  Cf = (float*)Cptr  + (size_t)blockIdx.z * sC;
    bf16_t* Cb = (bf16_t*)Cptr + (size_t)blockIdx.z * sC;
    #pragma unroll
    for (int mt = 0; mt < 4; ++mt) {
        const int mb = m0 + wm + mt * 16 + ((lane >> 4) ? 8 : 0);
        #pragma unroll
        for (int nt = 0; nt < 4; ++nt) {
            const int gn = n0 + wn + nt * 16 + l15;
            const float bval = HAS_BIAS ? bias[gn] : 0.0f;
            #pragma unroll
            for (int r = 0; r < 8; ++r) {
                const float val = acc[mt][nt][r] * scale + bval;
                const int gm = mb + r;
                if (OUT_F32)     Cf[(size_t)gm * ldc + gn] = val;
                else if (OUT_TR) Cb[(size_t)gn * ldc + gm] = (bf16_t)val;
                else             Cb[(size_t)gm * ldc + gn] = (bf16_t)val;
            }
        }
    }
}

// ---------------------------------------------------------------------------
// Row softmax over length-2048 rows, in place (f32) + bf16 copy for ctx GEMM.
// ---------------------------------------------------------------------------
__global__ __launch_bounds__(256)
void softmax_rows(float* __restrict__ scores, bf16_t* __restrict__ wbf)
{
    const size_t row = blockIdx.x;
    float* p = scores + row * (size_t)SEQ;
    __shared__ float red[8];
    const int tid = threadIdx.x;

    float4 va = *(const float4*)(p + tid * 4);
    float4 vb = *(const float4*)(p + 1024 + tid * 4);
    float vals[8] = { va.x, va.y, va.z, va.w, vb.x, vb.y, vb.z, vb.w };

    float lmax = -3.402823466e38f;
    #pragma unroll
    for (int i = 0; i < 8; ++i) lmax = fmaxf(lmax, vals[i]);
    #pragma unroll
    for (int off = 16; off; off >>= 1)
        lmax = fmaxf(lmax, __shfl_xor(lmax, off, 32));
    if ((tid & 31) == 0) red[tid >> 5] = lmax;
    __syncthreads();
    float rmax = red[0];
    #pragma unroll
    for (int j = 1; j < 8; ++j) rmax = fmaxf(rmax, red[j]);
    __syncthreads();

    float lsum = 0.0f;
    #pragma unroll
    for (int i = 0; i < 8; ++i) { vals[i] = __expf(vals[i] - rmax); lsum += vals[i]; }
    #pragma unroll
    for (int off = 16; off; off >>= 1) lsum += __shfl_xor(lsum, off, 32);
    if ((tid & 31) == 0) red[tid >> 5] = lsum;
    __syncthreads();
    float rsum = 0.0f;
    #pragma unroll
    for (int j = 0; j < 8; ++j) rsum += red[j];
    const float inv = 1.0f / rsum;

    #pragma unroll
    for (int i = 0; i < 8; ++i) vals[i] *= inv;
    *(float4*)(p + tid * 4)        = make_float4(vals[0], vals[1], vals[2], vals[3]);
    *(float4*)(p + 1024 + tid * 4) = make_float4(vals[4], vals[5], vals[6], vals[7]);
    bf16_t* wb = wbf + row * (size_t)SEQ;
    v4bf o0, o1;
    o0[0]=(bf16_t)vals[0]; o0[1]=(bf16_t)vals[1]; o0[2]=(bf16_t)vals[2]; o0[3]=(bf16_t)vals[3];
    o1[0]=(bf16_t)vals[4]; o1[1]=(bf16_t)vals[5]; o1[2]=(bf16_t)vals[6]; o1[3]=(bf16_t)vals[7];
    *(v4bf*)(wb + tid * 4)        = o0;
    *(v4bf*)(wb + 1024 + tid * 4) = o1;
}

// Straight fp32 -> bf16 convert, 8 elems/thread (b128 in, b128 out).
__global__ __launch_bounds__(256)
void cvt_f32_to_bf16_v8(const float* __restrict__ s, bf16_t* __restrict__ d)
{
    const size_t i = ((size_t)blockIdx.x * 256 + threadIdx.x) * 8;
    float4 f0 = *(const float4*)(s + i);
    float4 f1 = *(const float4*)(s + i + 4);
    v8bf o;
    o[0]=(bf16_t)f0.x; o[1]=(bf16_t)f0.y; o[2]=(bf16_t)f0.z; o[3]=(bf16_t)f0.w;
    o[4]=(bf16_t)f1.x; o[5]=(bf16_t)f1.y; o[6]=(bf16_t)f1.z; o[7]=(bf16_t)f1.w;
    *(v8bf*)(d + i) = o;
}

// Transpose + convert: W[K][N] f32 -> Wt[N][K] bf16 (32x32 LDS tiles).
__global__ __launch_bounds__(256)
void transpose_cvt(const float* __restrict__ src, bf16_t* __restrict__ dst,
                   int K, int N)
{
    __shared__ float t[32][33];
    const int k0 = blockIdx.y * 32, n0 = blockIdx.x * 32;
    const int tx = threadIdx.x & 31, ty = threadIdx.x >> 5;  // 32 x 8
    #pragma unroll
    for (int i = 0; i < 32; i += 8)
        t[ty + i][tx] = src[(size_t)(k0 + ty + i) * N + n0 + tx];
    __syncthreads();
    #pragma unroll
    for (int i = 0; i < 32; i += 8)
        dst[(size_t)(n0 + ty + i) * K + k0 + tx] = (bf16_t)t[tx][ty + i];
}

// ---------------------------------------------------------------------------
extern "C" void kernel_launch(void* const* d_in, const int* in_sizes, int n_in,
                              void* d_out, int out_size, void* d_ws, size_t ws_size,
                              hipStream_t stream)
{
    const float* q  = (const float*)d_in[0];
    const float* k  = (const float*)d_in[1];
    const float* v  = (const float*)d_in[2];
    const float* Wq = (const float*)d_in[3];
    const float* bq = (const float*)d_in[4];
    const float* Wk = (const float*)d_in[5];
    const float* bk = (const float*)d_in[6];
    const float* Wv = (const float*)d_in[7];
    const float* bv = (const float*)d_in[8];
    const float* Wo = (const float*)d_in[9];
    const float* bo = (const float*)d_in[10];

    const int B = BATCH, S = SEQ, D = DMODEL;
    float* outp = (float*)d_out;                      // (B,S,D) f32
    float* wts  = outp + (size_t)B * S * D;           // (B,S,S) f32 output + scratch

    char* ws = (char*)d_ws;
    size_t off = 0;
    bf16_t* WqT = (bf16_t*)(ws + off); off += (size_t)D * D * 2;   // [N][K]
    bf16_t* WkT = (bf16_t*)(ws + off); off += (size_t)D * D * 2;
    bf16_t* WvT = (bf16_t*)(ws + off); off += (size_t)D * D * 2;
    bf16_t* WoT = (bf16_t*)(ws + off); off += (size_t)D * D * 2;
    bf16_t* qb  = (bf16_t*)(ws + off); off += (size_t)B * S * D * 2;
    bf16_t* kb  = (bf16_t*)(ws + off); off += (size_t)B * S * D * 2;
    bf16_t* vb  = (bf16_t*)(ws + off); off += (size_t)B * S * D * 2;
    bf16_t* Qb  = (bf16_t*)(ws + off); off += (size_t)B * S * D * 2;
    bf16_t* Kb  = (bf16_t*)(ws + off); off += (size_t)B * S * D * 2;
    bf16_t* Vt  = (bf16_t*)(ws + off); off += (size_t)B * S * D * 2;  // [b][d][s]
    bf16_t* Wsm = (bf16_t*)(ws + off); off += (size_t)B * S * S * 2;  // bf16 weights
    bf16_t* Ctx = qb;   // reuse: qb dead after projections finish

    const dim3 blk256(256);
    const dim3 blkG(128);
    const dim3 gT(D / 32, D / 32);                    // (32,32)
    transpose_cvt<<<gT, blk256, 0, stream>>>(Wq, WqT, D, D);
    transpose_cvt<<<gT, blk256, 0, stream>>>(Wk, WkT, D, D);
    transpose_cvt<<<gT, blk256, 0, stream>>>(Wv, WvT, D, D);
    transpose_cvt<<<gT, blk256, 0, stream>>>(Wo, WoT, D, D);

    const int nConv = (B * S * D) / (256 * 8);        // 4096 blocks
    cvt_f32_to_bf16_v8<<<nConv, blk256, 0, stream>>>(q, qb);
    cvt_f32_to_bf16_v8<<<nConv, blk256, 0, stream>>>(k, kb);
    cvt_f32_to_bf16_v8<<<nConv, blk256, 0, stream>>>(v, vb);

    // Projections: per batch (S x D) = (S x D) @ W^T + bias
    const dim3 gProj(D / 128, S / 128, B);            // (8, 16, 4)
    gemm_wmma_bf16<false, false, true><<<gProj, blkG, 0, stream>>>(
        qb, WqT, bq, Qb, D, D, D, D, (long)S * D, 0, (long)S * D, 1.0f);
    gemm_wmma_bf16<false, false, true><<<gProj, blkG, 0, stream>>>(
        kb, WkT, bk, Kb, D, D, D, D, (long)S * D, 0, (long)S * D, 1.0f);
    gemm_wmma_bf16<false, true, true><<<gProj, blkG, 0, stream>>>(
        vb, WvT, bv, Vt, D, D, D, S, (long)S * D, 0, (long)D * S, 1.0f);  // -> Vt[d][s]

    // Head-averaged scores: (1/(H*sqrt(hd))) * Q @ K^T -> f32 in weights region
    const dim3 gScore(S / 128, S / 128, B);           // (16, 16, 4)
    gemm_wmma_bf16<true, false, false><<<gScore, blkG, 0, stream>>>(
        Qb, Kb, nullptr, wts, D, D, D, S,
        (long)S * D, (long)S * D, (long)S * S, 1.0f / 128.0f);

    softmax_rows<<<B * S, blk256, 0, stream>>>(wts, Wsm);

    // ctx = weights @ Vproj  (B operand = Vt[d][s], already [N][K])
    const dim3 gCtx(D / 128, S / 128, B);             // (8, 16, 4)
    gemm_wmma_bf16<false, false, false><<<gCtx, blkG, 0, stream>>>(
        Wsm, Vt, nullptr, Ctx, S, S, S, D,
        (long)S * S, (long)D * S, (long)S * D, 1.0f);

    // out = ctx @ Wo + bo -> f32
    gemm_wmma_bf16<true, false, true><<<gProj, blkG, 0, stream>>>(
        Ctx, WoT, bo, outp, D, D, D, D, (long)S * D, 0, (long)S * D, 1.0f);
}
